// HAN_60430189854755
// MI455X (gfx1250) — matrix operand order repaired and checked by reading, not compiled, not run on hardware
//
#include <hip/hip_runtime.h>
#include <hip/hip_bf16.h>

// ---------------------------------------------------------------------------
// HAN (2-layer hetero-GAT) for MI455X / gfx1250.
//  - Dense projections X@W use v_wmma_f32_16x16x32_f16 (f16 in, f32 acc),
//    64x128 block tile, B staged transposed in LDS so both A and B fragment
//    builds are contiguous -> ds_load_b128 pairs; B fragment reused across
//    4 row subtiles; uniform fast-path store epilogue.
//  - Attention logits el/er computed exactly in f32 via folded weights W·a.
//  - Edge softmax/aggregation: 3 bandwidth-bound passes with atomics; the
//    heavy scatter pass gathers float4 per thread (L2-resident feat arrays).
// ---------------------------------------------------------------------------

typedef _Float16 v16h __attribute__((ext_vector_type(16)));
typedef float    v8f  __attribute__((ext_vector_type(8)));

#define NHEAD 4
#define HF    128   // H * F (both layers: 4*32)
#define FOUT  32

// ---- monotonic float <-> uint encoding for atomicMax on floats -------------
__device__ __forceinline__ unsigned f2ord(float f) {
    unsigned u = __float_as_uint(f);
    return (u & 0x80000000u) ? ~u : (u | 0x80000000u);
}
__device__ __forceinline__ float ord2f(unsigned u) {
    return (u & 0x80000000u) ? __uint_as_float(u ^ 0x80000000u)
                             : __uint_as_float(~u);
}

// ---- fills ------------------------------------------------------------------
__global__ void fill_f32_kernel(float* p, float v, long n) {
    long i = (long)blockIdx.x * blockDim.x + threadIdx.x;
    if (i < n) p[i] = v;
}
__global__ void fill_u32_kernel(unsigned* p, unsigned v, long n) {
    long i = (long)blockIdx.x * blockDim.x + threadIdx.x;
    if (i < n) p[i] = v;
}

// ---- fold weights: v[k,h] = sum_f W[k, h*32+f] * a[h*32+f]  (K*4 <= 256) ----
__global__ void fold_weights_kernel(const float* __restrict__ W,
                                    const float* __restrict__ al,
                                    const float* __restrict__ ar,
                                    float* __restrict__ vl,
                                    float* __restrict__ vr, int K) {
    int tid = threadIdx.x;
    if (tid < K * NHEAD) {
        int k = tid >> 2, h = tid & 3;
        float sl = 0.f, sr = 0.f;
        for (int f = 0; f < FOUT; ++f) {
            float w = W[k * HF + h * FOUT + f];
            sl += w * al[h * FOUT + f];
            sr += w * ar[h * FOUT + f];
        }
        vl[tid] = sl;
        vr[tid] = sr;
    }
}

// ---- node scores: out[n,h] = sum_k X[n,k] * v[k,h] --------------------------
__global__ void node_scores_kernel(const float* __restrict__ X,
                                   const float* __restrict__ v,
                                   float* __restrict__ out, int N, int K) {
    int tid = blockIdx.x * blockDim.x + threadIdx.x;
    if (tid >= N * NHEAD) return;
    int n = tid >> 2, h = tid & 3;
    float s = 0.f;
    for (int k = 0; k < K; ++k) s += X[n * K + k] * v[k * NHEAD + h];
    out[tid] = s;
}

// ---- WMMA GEMM: feat[N,128] = f16(X[N,K]) @ f16(W[K,128]), f32 accumulate --
// Block = 256 threads = 8 waves. Block tile: 64 rows x 128 cols.
// Each wave owns one 16-wide column strip and 4 row subtiles (4 f32 accs),
// reusing one B fragment across 4 A fragments per 32-deep k-step.
// W is staged TRANSPOSED in LDS so each lane's B fragment is two contiguous
// 16B runs (ds_load_b128), matching the A fragment pattern.
template <int K>
__global__ void gemm_feat_kernel(const float* __restrict__ X,
                                 const float* __restrict__ W,
                                 float* __restrict__ feat,
                                 int Nnodes) {
    __shared__ _Float16 sA[64 * K];    // 64 rows x K
    __shared__ _Float16 sBt[HF * K];   // transposed: [col][k]

    const int t = threadIdx.x;
    const int row0 = blockIdx.x * 64;

    __builtin_prefetch(W, 0, 1);   // global_prefetch_b8

    // stage A tile (zero-pad OOB rows)
    for (int i = t; i < 64 * K; i += 256) {
        int r = i / K, k = i - r * K;
        int gr = row0 + r;
        float v = (gr < Nnodes) ? X[(long)gr * K + k] : 0.0f;
        sA[i] = (_Float16)v;
    }
    // stage W transposed: sBt[c*K + k] = W[k*128 + c]
    for (int i = t; i < K * HF; i += 256) {
        int k = i >> 7, c = i & (HF - 1);
        sBt[c * K + k] = (_Float16)W[i];
    }
    __syncthreads();

    const int lane    = t & 31;
    const int wave    = t >> 5;
    const int col0    = wave * 16;
    const int laneGrp = lane >> 4;    // 0: K-low half, 1: K-high half
    const int lr      = lane & 15;

    v8f acc[4];
#pragma unroll
    for (int m = 0; m < 4; ++m) acc[m] = v8f{0.f,0.f,0.f,0.f,0.f,0.f,0.f,0.f};

#pragma unroll
    for (int kb = 0; kb < K; kb += 32) {
        // B 32x16 f16 layout: lanes 0-15 hold K={kb..+7, kb+16..+23} (N=lane),
        // lanes 16-31 same, K offset +8. Two contiguous 8-half runs in sBt.
        v16h b;
        const int bbase = (col0 + lr) * K + kb + laneGrp * 8;
#pragma unroll
        for (int e = 0; e < 16; ++e) b[e] = sBt[bbase + (e & 7) + ((e >> 3) * 16)];

#pragma unroll
        for (int m = 0; m < 4; ++m) {
            // A 16x32 f16 layout: lanes 0-15 row M=lane, K={kb..+7, kb+16..+23};
            // lanes 16-31 same rows, K offset +8. Two contiguous 8-half runs.
            v16h a;
            const int arow = (m * 16 + lr) * K + kb + laneGrp * 8;
#pragma unroll
            for (int e = 0; e < 16; ++e)
                a[e] = sA[arow + (e & 7) + ((e >> 3) * 16)];
            acc[m] = __builtin_amdgcn_wmma_f32_16x16x32_f16(
                         /*neg_a=*/false, a, /*neg_b=*/false, b,
                         /*c_mod=*/(short)0, acc[m],
                         /*reuse_a=*/false, /*reuse_b=*/false);
        }
    }

    // D 16x16 f32 layout: element r -> M = r + laneGrp*8, N = lr
    // Uniform fast path: every block except the last has a full 64-row tile.
    if (row0 + 64 <= Nnodes) {
#pragma unroll
        for (int m = 0; m < 4; ++m) {
#pragma unroll
            for (int r = 0; r < 8; ++r) {
                int gr = row0 + m * 16 + r + laneGrp * 8;
                feat[(long)gr * HF + col0 + lr] = acc[m][r];
            }
        }
    } else {
#pragma unroll
        for (int m = 0; m < 4; ++m) {
#pragma unroll
            for (int r = 0; r < 8; ++r) {
                int gr = row0 + m * 16 + r + laneGrp * 8;
                if (gr < Nnodes) feat[(long)gr * HF + col0 + lr] = acc[m][r];
            }
        }
    }
}

// ---- edge pass 1: segment max of leaky_relu(el[src]+er[dst]) ---------------
__global__ void edge_max_kernel(const int* __restrict__ src,
                                const int* __restrict__ dst,
                                const float* __restrict__ el,
                                const float* __restrict__ er,
                                unsigned* __restrict__ mbits, long E) {
    long tid = (long)blockIdx.x * blockDim.x + threadIdx.x;
    if (tid >= E * NHEAD) return;
    long e = tid >> 2;
    int  h = (int)(tid & 3);
    float s = el[(long)src[e] * NHEAD + h] + er[(long)dst[e] * NHEAD + h];
    s = (s > 0.f) ? s : 0.2f * s;
    atomicMax(&mbits[(long)dst[e] * NHEAD + h], f2ord(s));
}

// ---- edge pass 2: denom[dst,h] += exp(s - m[dst,h]) ------------------------
__global__ void edge_denom_kernel(const int* __restrict__ src,
                                  const int* __restrict__ dst,
                                  const float* __restrict__ el,
                                  const float* __restrict__ er,
                                  const unsigned* __restrict__ mbits,
                                  float* __restrict__ denom, long E) {
    long tid = (long)blockIdx.x * blockDim.x + threadIdx.x;
    if (tid >= E * NHEAD) return;
    long e = tid >> 2;
    int  h = (int)(tid & 3);
    long di = (long)dst[e] * NHEAD + h;
    float s = el[(long)src[e] * NHEAD + h] + er[di];
    s = (s > 0.f) ? s : 0.2f * s;
    float w = __expf(s - ord2f(mbits[di]));
    atomicAdd(&denom[di], w);
}

// ---- edge pass 3: agg[dst, c..c+3] += (w/denom) * feat_src[src, c..c+3] ----
// One thread per (edge, 4-column group): float4 gather from feat_src.
__global__ void edge_agg_kernel(const int* __restrict__ src,
                                const int* __restrict__ dst,
                                const float* __restrict__ el,
                                const float* __restrict__ er,
                                const unsigned* __restrict__ mbits,
                                const float* __restrict__ denom,
                                const float* __restrict__ featS,
                                float* __restrict__ agg, long E) {
    long tid = (long)blockIdx.x * blockDim.x + threadIdx.x;
    if (tid >= E * (HF / 4)) return;
    long e = tid >> 5;              // HF/4 == 32 groups
    int  q = (int)(tid & 31);
    int  c = q * 4;
    int  h = c >> 5;                // head of this column group
    long di = (long)dst[e] * NHEAD + h;
    float s = el[(long)src[e] * NHEAD + h] + er[di];
    s = (s > 0.f) ? s : 0.2f * s;
    float a = __expf(s - ord2f(mbits[di])) / denom[di];
    const float4 v = *(const float4*)&featS[(long)src[e] * HF + c];
    float* dstp = &agg[(long)dst[e] * HF + c];
    atomicAdd(dstp + 0, a * v.x);
    atomicAdd(dstp + 1, a * v.y);
    atomicAdd(dstp + 2, a * v.z);
    atomicAdd(dstp + 3, a * v.w);
}

// ---- finalize: h_out[n,f] = mean_h(agg[n,h*32+f] + sum_rel_biases) ---------
__global__ void finalize_kernel(const float* __restrict__ agg,
                                const float* __restrict__ b0,
                                const float* __restrict__ b1,
                                const float* __restrict__ b2,
                                float* __restrict__ hout, int N) {
    int tid = blockIdx.x * blockDim.x + threadIdx.x;
    if (tid >= N * FOUT) return;
    int n = tid / FOUT, f = tid - n * FOUT;
    float s = 0.f;
    for (int h = 0; h < NHEAD; ++h) {
        int c = h * FOUT + f;
        float bb = b0[c];
        if (b1) bb += b1[c];
        if (b2) bb += b2[c];
        s += agg[(long)n * HF + c] + bb;
    }
    hout[tid] = s * (1.0f / NHEAD);
}

// ---- dot predictor ----------------------------------------------------------
__global__ void dot_kernel(const float* __restrict__ ha,
                           const float* __restrict__ ht,
                           const int* __restrict__ s,
                           const int* __restrict__ d,
                           float* __restrict__ out, long E) {
    long e = (long)blockIdx.x * blockDim.x + threadIdx.x;
    if (e >= E) return;
    const float* pa = ha + (long)s[e] * FOUT;
    const float* pt = ht + (long)d[e] * FOUT;
    float acc = 0.f;
#pragma unroll
    for (int f = 0; f < FOUT; ++f) acc += pa[f] * pt[f];
    out[e] = acc;
}

// ===========================================================================
extern "C" void kernel_launch(void* const* d_in, const int* in_sizes, int n_in,
                              void* d_out, int out_size, void* d_ws, size_t ws_size,
                              hipStream_t stream) {
    (void)in_sizes; (void)n_in; (void)out_size; (void)ws_size;

    // ---- problem constants ----
    const int NU = 40000, NA = 80000, NT = 15000;
    const long E_UA = 800000, E_UU = 400000, E_AT = 600000;
    const int cnt[3] = {NU, NA, NT};           // 0=user 1=artist 2=tag

    // relations in reference order: ua, au, uu0, uu1, at, ta
    const int  relSrcT[6] = {0, 1, 0, 0, 1, 2};
    const int  relDstT[6] = {1, 0, 0, 0, 2, 1};
    const long relE[6]    = {E_UA, E_UA, E_UU, E_UU, E_AT, E_AT};
    const int  relSrcIn[6] = {51, 53, 55, 57, 59, 61};
    const int  relDstIn[6] = {52, 54, 56, 58, 60, 62};
    // param flatten: layers sorted (layer1,layer2); rels sorted
    // (at,au,ta,ua,uu0,uu1); leaves sorted (W,al,ar,b)
    const int relSorted[6] = {3, 1, 4, 5, 0, 2};  // ua,au,uu0,uu1,at,ta

    // ---- carve workspace ----
    size_t off = 0;
    auto alloc = [&](size_t bytes) -> char* {
        char* p = (char*)d_ws + off;
        off = (off + bytes + 255) & ~(size_t)255;
        return p;
    };
    float*    featS[6]; float* elS[6]; float* erD[6];
    unsigned* mbits[6]; float* denom[6];
    float*    vl[6];    float* vr[6];
    for (int r = 0; r < 6; ++r) {
        featS[r] = (float*)alloc((size_t)cnt[relSrcT[r]] * HF * 4);
        elS[r]   = (float*)alloc((size_t)cnt[relSrcT[r]] * NHEAD * 4);
        erD[r]   = (float*)alloc((size_t)cnt[relDstT[r]] * NHEAD * 4);
        mbits[r] = (unsigned*)alloc((size_t)cnt[relDstT[r]] * NHEAD * 4);
        denom[r] = (float*)alloc((size_t)cnt[relDstT[r]] * NHEAD * 4);
        vl[r]    = (float*)alloc(64 * NHEAD * 4);
        vr[r]    = (float*)alloc(64 * NHEAD * 4);
    }
    float* agg[3]; float* outA[3]; float* outB[3];
    for (int ty = 0; ty < 3; ++ty) {
        agg[ty]  = (float*)alloc((size_t)cnt[ty] * HF * 4);
        outA[ty] = (float*)alloc((size_t)cnt[ty] * FOUT * 4);
        outB[ty] = (float*)alloc((size_t)cnt[ty] * FOUT * 4);
    }

    auto cdiv = [](long a, long b) -> long { return (a + b - 1) / b; };

    for (int L = 0; L < 2; ++L) {
        const int K = (L == 0) ? 64 : 32;
        const float* Xin[3];
        float* Out[3];
        for (int ty = 0; ty < 3; ++ty) {
            Xin[ty] = (L == 0) ? (const float*)d_in[48 + ty] : outA[ty];
            Out[ty] = (L == 0) ? outA[ty] : outB[ty];
        }

        // per-relation dense work
        for (int r = 0; r < 6; ++r) {
            const int st = relSrcT[r], dt = relDstT[r];
            const int base = 24 * L + 4 * relSorted[r];
            const float* W  = (const float*)d_in[base + 0];
            const float* al = (const float*)d_in[base + 1];
            const float* ar = (const float*)d_in[base + 2];

            fold_weights_kernel<<<1, 256, 0, stream>>>(W, al, ar, vl[r], vr[r], K);
            node_scores_kernel<<<cdiv((long)cnt[st] * NHEAD, 256), 256, 0, stream>>>(
                Xin[st], vl[r], elS[r], cnt[st], K);
            node_scores_kernel<<<cdiv((long)cnt[dt] * NHEAD, 256), 256, 0, stream>>>(
                Xin[dt], vr[r], erD[r], cnt[dt], K);
            if (K == 64)
                gemm_feat_kernel<64><<<cdiv(cnt[st], 64), 256, 0, stream>>>(
                    Xin[st], W, featS[r], cnt[st]);
            else
                gemm_feat_kernel<32><<<cdiv(cnt[st], 64), 256, 0, stream>>>(
                    Xin[st], W, featS[r], cnt[st]);

            long nd = (long)cnt[dt] * NHEAD;
            fill_u32_kernel<<<cdiv(nd, 256), 256, 0, stream>>>(mbits[r], 0u, nd);
            fill_f32_kernel<<<cdiv(nd, 256), 256, 0, stream>>>(denom[r], 0.f, nd);
        }
        for (int ty = 0; ty < 3; ++ty) {
            long n = (long)cnt[ty] * HF;
            fill_f32_kernel<<<cdiv(n, 256), 256, 0, stream>>>(agg[ty], 0.f, n);
        }

        // edge passes
        for (int r = 0; r < 6; ++r) {
            const int* s = (const int*)d_in[relSrcIn[r]];
            const int* d = (const int*)d_in[relDstIn[r]];
            const long E = relE[r];
            edge_max_kernel<<<cdiv(E * NHEAD, 256), 256, 0, stream>>>(
                s, d, elS[r], erD[r], mbits[r], E);
            edge_denom_kernel<<<cdiv(E * NHEAD, 256), 256, 0, stream>>>(
                s, d, elS[r], erD[r], mbits[r], denom[r], E);
            edge_agg_kernel<<<cdiv(E * (HF / 4), 256), 256, 0, stream>>>(
                s, d, elS[r], erD[r], mbits[r], denom[r], featS[r],
                agg[relDstT[r]], E);
        }

        // finalize per node type: user <- au,uu0,uu1; artist <- ua,ta; tag <- at
        const float* bU0 = (const float*)d_in[24 * L + 4 * relSorted[1] + 3];
        const float* bU1 = (const float*)d_in[24 * L + 4 * relSorted[2] + 3];
        const float* bU2 = (const float*)d_in[24 * L + 4 * relSorted[3] + 3];
        const float* bA0 = (const float*)d_in[24 * L + 4 * relSorted[0] + 3];
        const float* bA1 = (const float*)d_in[24 * L + 4 * relSorted[5] + 3];
        const float* bT0 = (const float*)d_in[24 * L + 4 * relSorted[4] + 3];
        finalize_kernel<<<cdiv((long)cnt[0] * FOUT, 256), 256, 0, stream>>>(
            agg[0], bU0, bU1, bU2, Out[0], cnt[0]);
        finalize_kernel<<<cdiv((long)cnt[1] * FOUT, 256), 256, 0, stream>>>(
            agg[1], bA0, bA1, nullptr, Out[1], cnt[1]);
        finalize_kernel<<<cdiv((long)cnt[2] * FOUT, 256), 256, 0, stream>>>(
            agg[2], bT0, nullptr, nullptr, Out[2], cnt[2]);
    }

    // predictor: pos on (at_src, at_dst), neg on (neg_src, neg_dst)
    float* outp = (float*)d_out;
    dot_kernel<<<cdiv(E_AT, 256), 256, 0, stream>>>(
        outB[1], outB[2], (const int*)d_in[59], (const int*)d_in[60], outp, E_AT);
    dot_kernel<<<cdiv(E_AT, 256), 256, 0, stream>>>(
        outB[1], outB[2], (const int*)d_in[63], (const int*)d_in[64],
        outp + E_AT, E_AT);
}
